// OpinionGuidedVAHead_67860483277732
// MI455X (gfx1250) — matrix operand order repaired and checked by reading, not compiled, not run on hardware
//
#include <hip/hip_runtime.h>

// ---------------- problem constants ----------------
#define BB 32
#define SS 512
#define HH 1024
#define QQ 64
#define VH 256
#define MROWS (BB * QQ)          // 2048
#define PAIRW (3 * HH)           // 3072

typedef __attribute__((ext_vector_type(16))) _Float16 v16h;
typedef __attribute__((ext_vector_type(8)))  _Float16 v8h;
typedef __attribute__((ext_vector_type(8)))  float    v8f;

static __device__ __forceinline__ float gelu_erf(float x) {
    return 0.5f * x * (1.0f + erff(x * 0.70710678118654752440f));
}
static __device__ __forceinline__ float sigmoidf_(float x) {
    return 1.0f / (1.0f + __expf(-x));
}

// ---------------- weight convert + transpose: W[K][N] f32 -> Wt[N][K] f16 ----
__global__ __launch_bounds__(256)
void k_wt_transpose(const float* __restrict__ W, _Float16* __restrict__ Wt,
                    int K, int N, int total) {
    int i = blockIdx.x * blockDim.x + threadIdx.x;
    if (i < total) {
        int k = i / N, n = i - k * N;
        Wt[(size_t)n * K + k] = (_Float16)W[i];
    }
}

// ---------------- span mean-pool -> f16 h_pair row [asp | opi | asp*opi] ----
__global__ __launch_bounds__(256)
void k_pool(const float* __restrict__ hidden, const int* __restrict__ spans,
            _Float16* __restrict__ hpair) {
    int row = blockIdx.x;                 // b*Q + q
    int b = row >> 6;                     // Q = 64
    const float* hb = hidden + (size_t)b * SS * HH;
    int4 sp = ((const int4*)spans)[row];  // sorted: x<=y<=z<=w
    bool va = (sp.x >= 2) && (sp.y >= sp.x);
    bool vo = (sp.z >= 2) && (sp.w >= sp.z);
    float inv_a = va ? 1.0f / (float)(sp.y - sp.x + 1) : 1.0f;
    float inv_o = vo ? 1.0f / (float)(sp.w - sp.z + 1) : 1.0f;
    _Float16* hp = hpair + (size_t)row * PAIRW;
    for (int j = threadIdx.x; j < HH; j += 256) {
        float aa, ao;
        if (va) { float s = 0.f; for (int r = sp.x; r <= sp.y; ++r) s += hb[r * HH + j]; aa = s * inv_a; }
        else    { aa = hb[HH + j]; }                 // fallback: hidden[:,1,:]
        if (vo) { float s = 0.f; for (int r = sp.z; r <= sp.w; ++r) s += hb[r * HH + j]; ao = s * inv_o; }
        else    { ao = hb[HH + j]; }
        hp[j]          = (_Float16)aa;
        hp[HH + j]     = (_Float16)ao;
        hp[2 * HH + j] = (_Float16)(aa * ao);
    }
}

// ---- WMMA GEMM, 16x64 strip per wave (4 accumulators, N-register tiling) ---
// out = GELU(A[M,K](lda) x W + bias), f16 out.  A row-major f16 (ld = lda),
// Wt is [N][K] f16 (pre-transposed weights), out row-major f16 [M, N].
// One wave per 16(M)x64(N) strip; 4 waves/block.  N must be a multiple of 64.
__global__ __launch_bounds__(128)
void k_wmma_gemm_gelu(const _Float16* __restrict__ A, int lda,
                      const _Float16* __restrict__ Wt, int K,
                      const float* __restrict__ bias,
                      _Float16* __restrict__ out, int N) {
    const int wave = threadIdx.x >> 5;
    const int lane = threadIdx.x & 31;
    const int ntiles = N >> 6;                       // 64-wide strips
    const int tile = blockIdx.x * 4 + wave;
    const int tm = (tile / ntiles) << 4;
    const int tn = (tile - (tile / ntiles) * ntiles) << 6;

    const int l15  = lane & 15;
    const int half = lane >> 4;                      // 0: lanes 0-15, 1: lanes 16-31
    // A frag (16x32 f16): lane holds row tm+l15; contiguous 16 halves on K (two 8-half loads)
    const _Float16* arow = A + (size_t)(tm + l15) * lda + half * 16;
    // B frags (32x16 f16): lane holds col tn+16n+l15; contiguous 16 halves on K
    const _Float16* brow = Wt + (size_t)(tn + l15) * K + half * 16;
    const size_t bstep = (size_t)16 * K;             // 16 columns of Wt

    v8f c0 = {}, c1 = {}, c2 = {}, c3 = {};
    for (int k0 = 0; k0 < K; k0 += 32) {
        v8h alo = *(const v8h*)(arow + k0);
        v8h ahi = *(const v8h*)(arow + k0 + 8);
        v16h a = __builtin_shufflevector(alo, ahi,
                 0,1,2,3,4,5,6,7,8,9,10,11,12,13,14,15);
        v16h b0 = *(const v16h*)(brow + k0);
        v16h b1 = *(const v16h*)(brow + bstep + k0);
        v16h b2 = *(const v16h*)(brow + 2 * bstep + k0);
        v16h b3 = *(const v16h*)(brow + 3 * bstep + k0);
        c0 = __builtin_amdgcn_wmma_f32_16x16x32_f16(false, a, false, b0, (short)0, c0, false, false);
        c1 = __builtin_amdgcn_wmma_f32_16x16x32_f16(false, a, false, b1, (short)0, c1, false, false);
        c2 = __builtin_amdgcn_wmma_f32_16x16x32_f16(false, a, false, b2, (short)0, c2, false, false);
        c3 = __builtin_amdgcn_wmma_f32_16x16x32_f16(false, a, false, b3, (short)0, c3, false, false);
    }
    // C/D layout: lane covers rows tm + half*8 + r, col tn + 16n + l15
    const int rbase = tm + half * 8;
    {
        const int col = tn + l15;       const float bv = bias[col];
#pragma unroll
        for (int r = 0; r < 8; ++r) out[(size_t)(rbase + r) * N + col] = (_Float16)gelu_erf(c0[r] + bv);
    }
    {
        const int col = tn + 16 + l15;  const float bv = bias[col];
#pragma unroll
        for (int r = 0; r < 8; ++r) out[(size_t)(rbase + r) * N + col] = (_Float16)gelu_erf(c1[r] + bv);
    }
    {
        const int col = tn + 32 + l15;  const float bv = bias[col];
#pragma unroll
        for (int r = 0; r < 8; ++r) out[(size_t)(rbase + r) * N + col] = (_Float16)gelu_erf(c2[r] + bv);
    }
    {
        const int col = tn + 48 + l15;  const float bv = bias[col];
#pragma unroll
        for (int r = 0; r < 8; ++r) out[(size_t)(rbase + r) * N + col] = (_Float16)gelu_erf(c3[r] + bv);
    }
}

// ---------------- tiny N=2 heads + gating + mask + output ------------------
__global__ __launch_bounds__(256)
void k_final(const _Float16* __restrict__ P1, const _Float16* __restrict__ R2,
             const float* __restrict__ pw2, const float* __restrict__ pb2,
             const float* __restrict__ rw3, const float* __restrict__ rb3,
             const float* __restrict__ gate_param,
             const float* __restrict__ qmask, float* __restrict__ out) {
    int idx = blockIdx.x * blockDim.x + threadIdx.x;
    if (idx >= MROWS) return;
    const _Float16* p = P1 + (size_t)idx * VH;
    const _Float16* r = R2 + (size_t)idx * (VH / 2);
    const bool keep = qmask[idx] >= 0.5f;
#pragma unroll
    for (int c = 0; c < 2; ++c) {
        float acc = pb2[c];
        for (int k = 0; k < VH; ++k) acc += (float)p[k] * pw2[2 * k + c];
        float prior = sigmoidf_(acc) * 8.0f + 1.0f;
        float acc2 = rb3[c];
        for (int k = 0; k < VH / 2; ++k) acc2 += (float)r[k] * rw3[2 * k + c];
        float delta = tanhf(acc2) * 4.0f;
        float gate = sigmoidf_(gate_param[c]);
        float vf = fminf(fmaxf(prior + gate * delta, 1.0f), 9.0f);
        out[2 * idx + c]             = keep ? vf    : 0.0f;   // va_final
        out[2 * MROWS + 2 * idx + c] = keep ? prior : 0.0f;   // va_prior
    }
}

// ---------------- workspace layout (bytes, all 256-aligned) ----------------
#define WS_HPAIR 0u
#define WS_PW1T  12582912u   // 2048*3072*2
#define WS_RW1T  13107200u   // + 256*1024*2
#define WS_RW2T  14680064u   // + 256*3072*2
#define WS_P1    14745600u   // + 128*256*2
#define WS_R1    15794176u   // + 2048*256*2
#define WS_R2    16842752u   // + 2048*256*2  (R2 ends at 17367040)

extern "C" void kernel_launch(void* const* d_in, const int* in_sizes, int n_in,
                              void* d_out, int out_size, void* d_ws, size_t ws_size,
                              hipStream_t stream) {
    (void)in_sizes; (void)n_in; (void)out_size; (void)ws_size;
    const float* hidden = (const float*)d_in[0];
    const int*   spans  = (const int*)  d_in[1];
    const float* qmask  = (const float*)d_in[2];
    const float* pw1 = (const float*)d_in[3];
    const float* pb1 = (const float*)d_in[4];
    const float* pw2 = (const float*)d_in[5];
    const float* pb2 = (const float*)d_in[6];
    const float* rw1 = (const float*)d_in[7];
    const float* rb1 = (const float*)d_in[8];
    const float* rw2 = (const float*)d_in[9];
    const float* rb2 = (const float*)d_in[10];
    const float* rw3 = (const float*)d_in[11];
    const float* rb3 = (const float*)d_in[12];
    const float* gate = (const float*)d_in[13];

    char* ws = (char*)d_ws;
    _Float16* hpair = (_Float16*)(ws + WS_HPAIR);
    _Float16* pw1t  = (_Float16*)(ws + WS_PW1T);
    _Float16* rw1t  = (_Float16*)(ws + WS_RW1T);
    _Float16* rw2t  = (_Float16*)(ws + WS_RW2T);
    _Float16* P1    = (_Float16*)(ws + WS_P1);
    _Float16* R1    = (_Float16*)(ws + WS_R1);
    _Float16* R2    = (_Float16*)(ws + WS_R2);

    // 1) weight transposes to [N][K] f16
    k_wt_transpose<<<(HH * VH + 255) / 256, 256, 0, stream>>>(pw1, pw1t, HH, VH, HH * VH);
    k_wt_transpose<<<(PAIRW * VH + 255) / 256, 256, 0, stream>>>(rw1, rw1t, PAIRW, VH, PAIRW * VH);
    k_wt_transpose<<<(VH * (VH / 2) + 255) / 256, 256, 0, stream>>>(rw2, rw2t, VH, VH / 2, VH * (VH / 2));

    // 2) span pooling -> f16 h_pair rows
    k_pool<<<MROWS, 256, 0, stream>>>(hidden, spans, hpair);

    // 3) prior L1: opi section of hpair (offset HH, lda=PAIRW), K=1024, N=256
    {
        int tiles = (MROWS / 16) * (VH / 64);       // 128 * 4 = 512
        k_wmma_gemm_gelu<<<tiles / 4, 128, 0, stream>>>(hpair + HH, PAIRW, pw1t, HH, pb1, P1, VH);
    }
    // 4) residual L1: full hpair, K=3072, N=256
    {
        int tiles = (MROWS / 16) * (VH / 64);       // 512
        k_wmma_gemm_gelu<<<tiles / 4, 128, 0, stream>>>(hpair, PAIRW, rw1t, PAIRW, rb1, R1, VH);
    }
    // 5) residual L2: K=256, N=128
    {
        int tiles = (MROWS / 16) * ((VH / 2) / 64); // 128 * 2 = 256
        k_wmma_gemm_gelu<<<tiles / 4, 128, 0, stream>>>(R1, VH, rw2t, VH, rb2, R2, VH / 2);
    }
    // 6) heads + gate + clip + mask
    k_final<<<(MROWS + 255) / 256, 256, 0, stream>>>(P1, R2, pw2, pb2, rw3, rb3, gate, qmask, (float*)d_out);
}
